// SGSBAttentionND_77043123355935
// MI455X (gfx1250) — matrix operand order repaired and checked by reading, not compiled, not run on hardware
//
#include <hip/hip_runtime.h>
#include <hip/hip_bf16.h>

typedef __attribute__((ext_vector_type(16))) __bf16 v16bf;
typedef __attribute__((ext_vector_type(8)))  float  v8f;

union FragBF { v16bf v; unsigned int u[8]; };

__device__ __forceinline__ float sigmoidf_(float x) { return 1.0f / (1.0f + __expf(-x)); }
__device__ __forceinline__ float siluf_(float x)    { return x * sigmoidf_(x); }

__device__ __forceinline__ unsigned short f2bf(float f) {
    unsigned int u = __float_as_uint(f);
    u = (u + 0x7fffu + ((u >> 16) & 1u)) >> 16;   // round-to-nearest-even
    return (unsigned short)u;
}

// async DMA global->LDS, 16B per lane; tracked by ASYNCcnt
__device__ __forceinline__ void async_copy_b128(unsigned lds_off, const void* gptr) {
    unsigned long long g = (unsigned long long)(uintptr_t)gptr;
    asm volatile("global_load_async_to_lds_b128 %0, %1, off"
                 :: "v"(lds_off), "v"(g) : "memory");
}
__device__ __forceinline__ void wait_async3() { asm volatile("s_wait_asynccnt 0x3" ::: "memory"); }
__device__ __forceinline__ void wait_async0() { asm volatile("s_wait_asynccnt 0x0" ::: "memory"); }

// ---------------------------------------------------------------- prep kernels
__global__ void cvt_bf16_kernel(const float* __restrict__ src,
                                unsigned short* __restrict__ dst, int n) {
    int i = blockIdx.x * blockDim.x + threadIdx.x;
    if (i < n) dst[i] = f2bf(src[i]);
}

__global__ void combine_bias_kernel(const float* __restrict__ bw,
                                    const float* __restrict__ bq,
                                    float* __restrict__ dst) {
    int i = blockIdx.x * blockDim.x + threadIdx.x;
    if (i < 304) dst[i] = (i < 48) ? bw[i] : bq[i - 48];
}

// ---------------------------------------------------------------- WMMA GEMM
// C[m,n] = act(sum_k A[m,k]*W[n,k] + bias[n]) ; A,W bf16 row-major.
// Block tile 64(M) x 128(N), K chunked by 32, double-buffered LDS fed by
// global_load_async_to_lds_b128.  8 waves: wave = (wm 0..3) x (wn 0..1),
// each wave owns a 16x64 strip = 4 WMMA tiles sharing one A fragment.
enum { EPI_SILU_F32 = 0, EPI_SILU_BF16 = 1, EPI_SIGMUL_BF16 = 2 };

#define LDSTR 40   // padded row stride in bf16 (80B: 16B-aligned, conflict-free frags)

__global__ __launch_bounds__(256)
void gemm_bf16_wmma(const unsigned short* __restrict__ A, int lda,
                    const unsigned short* __restrict__ W, int ldw,
                    const float* __restrict__ bias,
                    float* __restrict__ outF,
                    unsigned short* __restrict__ outB,
                    const float* __restrict__ aux,
                    int ldo, int M, int N, int K, int epi)
{
    __shared__ unsigned short As[2][64 * LDSTR];
    __shared__ unsigned short Bs[2][128 * LDSTR];

    const int t    = threadIdx.x;
    const int m0   = blockIdx.y * 64;
    const int n0   = blockIdx.x * 128;
    const int wave = t >> 5, lane = t & 31;
    const int wm   = wave >> 1;          // 0..3 : 16-row strip
    const int wn   = wave & 1;           // 0..1 : 64-col strip (4 tiles)
    const int half = lane >> 4;          // lane group
    const int lr   = lane & 15;

    // per-thread copy coordinates (8 bf16 = one b128 per issue)
    const int cm = t >> 2;               // 0..63
    const int cc = (t & 3) * 8;          // 0,8,16,24
    int gmA = m0 + cm; if (gmA >= M) gmA = M - 1;
    int gn0 = n0 + cm;       if (gn0 >= N) gn0 = N - 1;
    int gn1 = n0 + cm + 64;  if (gn1 >= N) gn1 = N - 1;

    v8f acc[4] = {{}, {}, {}, {}};

    const int nk = K >> 5;
    auto issue = [&](int kt, int buf) {
        int k0 = kt << 5;
        async_copy_b128((unsigned)(uintptr_t)&As[buf][cm * LDSTR + cc],
                        A + (size_t)gmA * lda + k0 + cc);
        async_copy_b128((unsigned)(uintptr_t)&Bs[buf][cm * LDSTR + cc],
                        W + (size_t)gn0 * ldw + k0 + cc);
        async_copy_b128((unsigned)(uintptr_t)&Bs[buf][(cm + 64) * LDSTR + cc],
                        W + (size_t)gn1 * ldw + k0 + cc);
    };

    issue(0, 0);
    for (int kt = 0; kt < nk; ++kt) {
        const int buf = kt & 1;
        if (kt + 1 < nk) { issue(kt + 1, buf ^ 1); wait_async3(); }
        else             { wait_async0(); }
        __syncthreads();                         // tile kt resident in LDS (all waves)

        FragBF a;
#pragma unroll
        for (int j = 0; j < 8; ++j) {
            // 16-bit A 16x32 lane layout (ISA 7.12.2)
            int ka = (j < 4) ? (half * 8 + 2 * j) : (16 + half * 8 + 2 * (j - 4));
            a.u[j] = *(const unsigned int*)&As[buf][(wm * 16 + lr) * LDSTR + ka];
        }
#pragma unroll
        for (int nt = 0; nt < 4; ++nt) {
            FragBF b;
#pragma unroll
            for (int j = 0; j < 8; ++j) {
                // 16-bit B 32x16: same K order as A (K=8h+s for s<8, 16+8h+(s-8) for s>=8)
                int kb = (j < 4) ? (half * 8 + 2 * j) : (16 + half * 8 + 2 * (j - 4));
                b.u[j] = *(const unsigned int*)&Bs[buf][(wn * 64 + nt * 16 + lr) * LDSTR + kb];
            }
            acc[nt] = __builtin_amdgcn_wmma_f32_16x16x32_bf16(false, a.v, false, b.v,
                                                              (short)0, acc[nt], false, false);
        }
        __syncthreads();                         // all waves done reading buf
    }

#pragma unroll
    for (int nt = 0; nt < 4; ++nt) {
        int gn = n0 + wn * 64 + nt * 16 + lr;
        if (gn >= N) continue;
        float bv = bias ? bias[gn] : 0.0f;
#pragma unroll
        for (int r = 0; r < 8; ++r) {
            int gm = m0 + wm * 16 + half * 8 + r;    // C/D: VGPR r -> M = r + 8*half
            if (gm >= M) continue;
            float v = acc[nt][r] + bv;
            size_t o = (size_t)gm * ldo + gn;
            if (epi == EPI_SILU_F32)       { outF[o] = siluf_(v); }
            else if (epi == EPI_SILU_BF16) { outB[o] = f2bf(siluf_(v)); }
            else /* EPI_SIGMUL_BF16 */     { outB[o] = f2bf(aux[o] * sigmoidf_(v)); }
        }
    }
}

// ---------------------------------------------------------------- wave params
__global__ void wave_post_kernel(const float* __restrict__ wq,
                                 float* __restrict__ freqb,
                                 float* __restrict__ decayb,
                                 float* __restrict__ favgb,
                                 float* __restrict__ pavgb, int rows)
{
    int row = blockIdx.x * blockDim.x + threadIdx.x;
    if (row >= rows) return;
    const float* wp = wq + (size_t)row * 304;   // first 48 cols = silu(wave proj)
    float fs = 0.f, ps = 0.f;
#pragma unroll
    for (int h = 0; h < 16; ++h) {
        float f  = sigmoidf_(wp[h])      * 15.0f + 1.0f;   // freq in [1,16]
        float ph = tanhf(wp[16 + h])     * 16.0f;          // phase
        float dc = sigmoidf_(wp[32 + h]) * 9.5f + 0.5f;    // decay
        freqb[row * 16 + h]  = f;
        decayb[row * 16 + h] = dc;
        fs += f; ps += ph;
    }
    favgb[row] = fs * 0.0625f;
    pavgb[row] = ps * 0.0625f;
}

// ---------------------------------------------------------------- attention
__global__ __launch_bounds__(256)
void attn_kernel(const float* __restrict__ x,
                 const float* __restrict__ wq,      // [rows][304], cols 48.. = queries
                 const float* __restrict__ freqb,
                 const float* __restrict__ decayb,
                 const float* __restrict__ favgb,
                 const float* __restrict__ pavgb,
                 const float* __restrict__ wkey,    // [16]
                 float* __restrict__ attnF,
                 unsigned short* __restrict__ attnB)
{
    __shared__ float qs[256];
    __shared__ float freqs[16], decays[16], wk[16];
    __shared__ int   idxS[33];
    __shared__ float vmask[33];
    __shared__ float wgt[16][34];

    const int row = blockIdx.x;
    const int b   = row >> 11;           // / 2048
    const int l   = row & 2047;
    const int t   = threadIdx.x;

    qs[t] = wq[(size_t)row * 304 + 48 + t];
    if (t < 16) { freqs[t] = freqb[row * 16 + t];
                  decays[t] = decayb[row * 16 + t];
                  wk[t] = wkey[t]; }
    if (t < 33) {
        float favg = favgb[row], pavg = pavgb[row];
        float pos = (float)l + (float)(t - 16) * favg + pavg;
        int valid = (pos >= 0.0f) && (pos < 2048.0f);
        int idx = (int)pos;                           // trunc toward zero
        idx = idx < 0 ? 0 : (idx > 2047 ? 2047 : idx);
        idxS[t] = idx; vmask[t] = valid ? 1.0f : 0.0f;
    }
    __syncthreads();

    // logits[h][s] = 0.25 * sum_d q[h,d] * silu(|s-16|*freq[h]*wk[d])
    for (int p = t; p < 528; p += 256) {
        int h = p / 33, s = p - h * 33;
        float rel = fabsf((float)(s - 16)) * freqs[h];
        float av = 0.0f;
#pragma unroll
        for (int d = 0; d < 16; ++d) av += qs[h * 16 + d] * siluf_(rel * wk[d]);
        wgt[h][s] = (vmask[s] > 0.0f) ? (av * 0.25f) : -1.0e30f;
    }
    __syncthreads();

    if (t < 16) {   // masked softmax * envelope, renormalize
        int h = t;
        float mx = -3.4e38f;
        for (int s = 0; s < 33; ++s) mx = fmaxf(mx, wgt[h][s]);
        float sum = 0.0f;
        for (int s = 0; s < 33; ++s) { float e = __expf(wgt[h][s] - mx); wgt[h][s] = e; sum += e; }
        float inv = 1.0f / sum;
        float dcl = fmaxf(decays[h], 0.1f);
        float sum2 = 0.0f;
        for (int s = 0; s < 33; ++s) {
            float rel = fabsf((float)(s - 16)) * freqs[h];
            float w = wgt[h][s] * inv * __expf(-rel / dcl) * vmask[s];
            wgt[h][s] = w; sum2 += w;
        }
        float inv2 = 1.0f / (sum2 + 1.0e-8f);
        for (int s = 0; s < 33; ++s) wgt[h][s] *= inv2;
    }
    __syncthreads();

    const float* xb = x + (size_t)b * 2048 * 1024;
#pragma unroll
    for (int j = 0; j < 4; ++j) {
        int c = t + 256 * j;                 // coalesced channels
        int h = c >> 6;
        float av = 0.0f;
        for (int s = 0; s < 33; ++s)
            av += wgt[h][s] * xb[(size_t)idxS[s] * 1024 + c];
        size_t o = (size_t)row * 1024 + c;
        attnF[o] = av;
        attnB[o] = f2bf(av);
    }
}

// ---------------------------------------------------------------- launch
extern "C" void kernel_launch(void* const* d_in, const int* in_sizes, int n_in,
                              void* d_out, int out_size, void* d_ws, size_t ws_size,
                              hipStream_t stream) {
    const float* x       = (const float*)d_in[0];   // (2,2048,1024)
    const float* w_wave  = (const float*)d_in[1];   // (48,1024)
    const float* b_wave  = (const float*)d_in[2];
    const float* w_query = (const float*)d_in[3];   // (256,1024)
    const float* b_query = (const float*)d_in[4];
    const float* w_key   = (const float*)d_in[5];   // (16,1)
    const float* w_out   = (const float*)d_in[6];   // (1024,1024)
    const float* w_se1   = (const float*)d_in[7];   // (256,1024)
    const float* b_se1   = (const float*)d_in[8];
    const float* w_se2   = (const float*)d_in[9];   // (1024,256)
    const float* b_se2   = (const float*)d_in[10];

    const int ROWS = 4096;          // B*L
    char* ws = (char*)d_ws;
    size_t off = 0;
    auto alloc = [&](size_t bytes) { size_t o = off; off += (bytes + 255) & ~(size_t)255; return o; };

    unsigned short* x_bf    = (unsigned short*)(ws + alloc((size_t)ROWS * 1024 * 2));
    unsigned short* w1_bf   = (unsigned short*)(ws + alloc((size_t)304 * 1024 * 2));
    unsigned short* wse1_bf = (unsigned short*)(ws + alloc((size_t)256 * 1024 * 2));
    unsigned short* wse2_bf = (unsigned short*)(ws + alloc((size_t)1024 * 256 * 2));
    unsigned short* wout_bf = (unsigned short*)(ws + alloc((size_t)1024 * 1024 * 2));
    float*          bias1   = (float*)(ws + alloc(304 * 4));
    float*          wqbuf   = (float*)(ws + alloc((size_t)ROWS * 304 * 4));
    float*          freqb   = (float*)(ws + alloc((size_t)ROWS * 16 * 4));
    float*          decayb  = (float*)(ws + alloc((size_t)ROWS * 16 * 4));
    float*          favgb   = (float*)(ws + alloc((size_t)ROWS * 4));
    float*          pavgb   = (float*)(ws + alloc((size_t)ROWS * 4));
    float*          attnF   = (float*)(ws + alloc((size_t)ROWS * 1024 * 4));
    unsigned short* attnB   = (unsigned short*)(ws + alloc((size_t)ROWS * 1024 * 2));
    unsigned short* h_bf    = (unsigned short*)(ws + alloc((size_t)ROWS * 256 * 2));
    unsigned short* y_bf    = (unsigned short*)(ws + alloc((size_t)ROWS * 1024 * 2));
    (void)ws_size; (void)n_in; (void)in_sizes; (void)out_size;

    auto cgrid = [](int n) { return (n + 255) / 256; };

    // bf16 conversions (weights + activations)
    cvt_bf16_kernel<<<cgrid(ROWS * 1024), 256, 0, stream>>>(x, x_bf, ROWS * 1024);
    cvt_bf16_kernel<<<cgrid(48 * 1024),   256, 0, stream>>>(w_wave,  w1_bf,           48 * 1024);
    cvt_bf16_kernel<<<cgrid(256 * 1024),  256, 0, stream>>>(w_query, w1_bf + 48*1024, 256 * 1024);
    cvt_bf16_kernel<<<cgrid(256 * 1024),  256, 0, stream>>>(w_se1,   wse1_bf,         256 * 1024);
    cvt_bf16_kernel<<<cgrid(1024 * 256),  256, 0, stream>>>(w_se2,   wse2_bf,         1024 * 256);
    cvt_bf16_kernel<<<cgrid(1024 * 1024), 256, 0, stream>>>(w_out,   wout_bf,         1024 * 1024);
    combine_bias_kernel<<<2, 256, 0, stream>>>(b_wave, b_query, bias1);

    // GEMM1: wp+queries = silu(x @ [w_wave;w_query].T + b)  -> wqbuf [4096][304]
    gemm_bf16_wmma<<<dim3(3, 64), 256, 0, stream>>>(
        x_bf, 1024, w1_bf, 1024, bias1, wqbuf, nullptr, nullptr,
        304, ROWS, 304, 1024, EPI_SILU_F32);

    wave_post_kernel<<<cgrid(ROWS), 256, 0, stream>>>(wqbuf, freqb, decayb, favgb, pavgb, ROWS);

    // gather attention core -> attnF (f32) + attnB (bf16)
    attn_kernel<<<ROWS, 256, 0, stream>>>(x, wqbuf, freqb, decayb, favgb, pavgb,
                                          w_key, attnF, attnB);

    // SE1: h = silu(attn @ w_se1.T + b_se1)  -> bf16 [4096][256]
    gemm_bf16_wmma<<<dim3(2, 64), 256, 0, stream>>>(
        attnB, 1024, wse1_bf, 1024, b_se1, nullptr, h_bf, nullptr,
        256, ROWS, 256, 1024, EPI_SILU_BF16);

    // SE2: y = attn * sigmoid(h @ w_se2.T + b_se2)  -> bf16 [4096][1024]
    gemm_bf16_wmma<<<dim3(8, 64), 256, 0, stream>>>(
        h_bf, 256, wse2_bf, 256, b_se2, nullptr, y_bf, attnF,
        1024, ROWS, 1024, 256, EPI_SIGMUL_BF16);

    // OUT: d_out = silu(y @ w_out.T)  -> f32
    gemm_bf16_wmma<<<dim3(8, 64), 256, 0, stream>>>(
        y_bf, 1024, wout_bf, 1024, nullptr, (float*)d_out, nullptr, nullptr,
        1024, ROWS, 1024, 1024, EPI_SILU_F32);
}
